// Block_ViT_46755013984667
// MI455X (gfx1250) — matrix operand order repaired and checked by reading, not compiled, not run on hardware
//
#include <hip/hip_runtime.h>
#include <math.h>

// ---------------- problem constants ----------------
namespace {
constexpr int kB   = 4;      // batch
constexpr int kN   = 784;    // tokens
constexpr int kNp  = 800;    // tokens padded to multiple of BK (25*32)
constexpr int kH   = 4;      // heads
constexpr int kKV  = 960;    // concat channel dim
constexpr int kCs[4] = {64, 128, 256, 512};

constexpr int BM = 128, BN = 64, BK = 32;  // GEMM block tile
constexpr int LDSS = BK + 8;               // LDS row stride (ushort) = 80B
}

typedef __attribute__((ext_vector_type(16))) __bf16 bf16x16;
typedef __attribute__((ext_vector_type(8)))  float  f32x8;

__device__ __forceinline__ unsigned short f32_to_bf16_rne(float f) {
  union { float f; unsigned int u; } c; c.f = f;
  unsigned int u = c.u;
  return (unsigned short)((u + 0x7FFFu + ((u >> 16) & 1u)) >> 16);
}

__device__ __forceinline__ unsigned int pack_bf16x2(float lo, float hi) {
  return ((unsigned int)f32_to_bf16_rne(hi) << 16) | (unsigned int)f32_to_bf16_rne(lo);
}

// ---------------------------------------------------------------------------
// Batched GEMM:  Y[m,n] = alpha * sum_k A[m,k]*B[n,k]  (+bias)(+gelu)(+=C)
//   A: Abase + m*a_rs + k        (k contiguous, K % 32 == 0 required)
//   B: Bbase + n*b_rs + k        (k contiguous, N % 64 == 0 required)
//   C: Cbase + m*c_rs + n*c_cs   (c_cs != 1 => transposed store)
// Rows m >= M are clamped on load; stores happen for m < Mzero, with value
// zeroed for m in [M, Mzero) (used to zero-pad transposed Q/K outputs).
// batch z -> (ib = z / H2, ih = z % H2).
// flags: 1 = accumulate into C, 2 = add bias[n], 4 = exact GELU
//
// 128 threads = 4 waves; wave w computes rows [w*32, w*32+32) x all 64 cols:
// 2 A-frags x 4 B-frags = 8 WMMAs per wave per k-tile. Double-buffered LDS,
// one barrier per k-tile, global loads overlapped with WMMA.
// ---------------------------------------------------------------------------
struct GemmP {
  const float* A; const float* Bm; const float* bias; float* C;
  int M, N, K, Mzero;
  long a_rs, b_rs, c_rs, c_cs;
  long a_sb, a_sh, b_sb, b_sh, c_sb, c_sh;
  int H2;
  float alpha;
  int flags;
};

__global__ __launch_bounds__(128)
void gemm_bf16_wmma(GemmP p) {
  __shared__ unsigned short As[2][BM][LDSS];   // 2 x 128 x 40 ushort = 20 KB
  __shared__ unsigned short Bs[2][BN][LDSS];   // 2 x  64 x 40 ushort = 10 KB

  const int z  = blockIdx.z;
  const int ib = z / p.H2;
  const int ih = z % p.H2;
  const float* Ab = p.A  + (long)ib * p.a_sb + (long)ih * p.a_sh;
  const float* Bb = p.Bm + (long)ib * p.b_sb + (long)ih * p.b_sh;
  float*       Cb = p.C  + (long)ib * p.c_sb + (long)ih * p.c_sh;

  const int tile_m = blockIdx.y * BM;
  const int tile_n = blockIdx.x * BN;
  const int tid  = threadIdx.x;
  const int lane = tid & 31;
  const int wave = tid >> 5;
  const int wm   = wave * 32;        // wave row origin within the 128-row tile
  const int hlf  = lane >> 4;        // 0: lanes 0-15, 1: lanes 16-31
  const int l16  = lane & 15;

  // Per-thread staging slots: A = 8 float4 (128x8), B = 4 float4 (64x8).
  const float* aP[8]; const float* bP[4];
  #pragma unroll
  for (int i = 0; i < 8; ++i) {
    const int idx = tid + i * 128;   // 0..1023
    const int r   = idx >> 3;        // 0..127
    const int c4  = idx & 7;         // 0..7
    int gm = tile_m + r;
    gm = (gm < p.M) ? gm : (p.M - 1);          // branch-free row clamp
    aP[i] = Ab + (long)gm * p.a_rs + c4 * 4;
  }
  #pragma unroll
  for (int i = 0; i < 4; ++i) {
    const int idx = tid + i * 128;   // 0..511
    const int r   = idx >> 3;        // 0..63
    const int c4  = idx & 7;
    const int gn  = tile_n + r;                // N % 64 == 0: always in range
    bP[i] = Bb + (long)gn * p.b_rs + c4 * 4;
  }

  f32x8 acc[2][4];
  #pragma unroll
  for (int i = 0; i < 2; ++i)
    #pragma unroll
    for (int j = 0; j < 4; ++j)
      #pragma unroll
      for (int r = 0; r < 8; ++r) acc[i][j][r] = 0.0f;

  float4 av[8], bv[4];

  auto load_tile = [&]() {
    #pragma unroll
    for (int i = 0; i < 8; ++i) { av[i] = *(const float4*)aP[i]; aP[i] += BK; }
    #pragma unroll
    for (int i = 0; i < 4; ++i) { bv[i] = *(const float4*)bP[i]; bP[i] += BK; }
    __builtin_prefetch(aP[0], 0, 1);           // next k-tile: global_prefetch_b8
    __builtin_prefetch(bP[0], 0, 1);
  };

  auto store_tile = [&](int buf) {
    #pragma unroll
    for (int i = 0; i < 8; ++i) {
      const int idx = tid + i * 128;
      const int r = idx >> 3, c4 = idx & 7;
      uint2 w;
      w.x = pack_bf16x2(av[i].x, av[i].y);
      w.y = pack_bf16x2(av[i].z, av[i].w);
      *(uint2*)&As[buf][r][c4 * 4] = w;        // ds_store_b64
    }
    #pragma unroll
    for (int i = 0; i < 4; ++i) {
      const int idx = tid + i * 128;
      const int r = idx >> 3, c4 = idx & 7;
      uint2 w;
      w.x = pack_bf16x2(bv[i].x, bv[i].y);
      w.y = pack_bf16x2(bv[i].z, bv[i].w);
      *(uint2*)&Bs[buf][r][c4 * 4] = w;
    }
  };

  auto compute_tile = [&](int buf) {
    // WMMA fragments per documented CDNA5 VGPR layouts.
    union Frag { unsigned short u[16]; bf16x16 v; };
    Frag afr[2], bfr[4];
    #pragma unroll
    for (int i = 0; i < 2; ++i) {
      // A 16x32: lane row m=l16; j<8 -> k=hlf*8+j ; j>=8 -> k=16+hlf*8+(j-8)
      const unsigned short* rowA = &As[buf][wm + i * 16 + l16][0];
      #pragma unroll
      for (int j = 0; j < 8; ++j) {
        afr[i].u[j]     = rowA[hlf * 8 + j];
        afr[i].u[8 + j] = rowA[16 + hlf * 8 + j];
      }
    }
    #pragma unroll
    for (int j = 0; j < 4; ++j) {
      // B 32x16: lane col n=l16; t<8 -> k=hlf*8+t ; t>=8 -> k=16+hlf*8+(t-8)
      const unsigned short* rowB = &Bs[buf][j * 16 + l16][0];
      #pragma unroll
      for (int t = 0; t < 16; ++t) bfr[j].u[t] = rowB[(t >> 3) * 16 + hlf * 8 + (t & 7)];
    }
    #pragma unroll
    for (int i = 0; i < 2; ++i)
      #pragma unroll
      for (int j = 0; j < 4; ++j)
        acc[i][j] = __builtin_amdgcn_wmma_f32_16x16x32_bf16(
            false, afr[i].v, false, bfr[j].v, (short)0, acc[i][j], false, false);
  };

  // Pipeline: prologue fills buffer 0; per iteration issue next tile's global
  // loads, WMMA on current buffer while they are in flight, store into the
  // other buffer, single barrier.
  const int kTiles = p.K / BK;   // K % 32 == 0 by construction
  load_tile();
  store_tile(0);
  __syncthreads();
  for (int kt = 0; kt < kTiles; ++kt) {
    const int cur = kt & 1;
    const bool more = (kt + 1 < kTiles);
    if (more) load_tile();
    compute_tile(cur);
    if (more) store_tile(cur ^ 1);
    __syncthreads();
  }

  // Epilogue. D layout: lane l16 = n; VGPR r -> m = hlf*8 + r.
  #pragma unroll
  for (int i = 0; i < 2; ++i) {
    #pragma unroll
    for (int j = 0; j < 4; ++j) {
      const int n = tile_n + j * 16 + l16;
      #pragma unroll
      for (int r = 0; r < 8; ++r) {
        const int m = tile_m + wm + i * 16 + hlf * 8 + r;
        if (m < p.Mzero) {
          float v = (m < p.M) ? p.alpha * acc[i][j][r] : 0.0f;
          if (p.flags & 2) v += p.bias[n];
          if (p.flags & 4) v = 0.5f * v * (1.0f + erff(v * 0.70710678118654752f));
          const long co = (long)m * p.c_rs + (long)n * p.c_cs;
          if (p.flags & 1) v += Cb[co];
          Cb[co] = v;
        }
      }
    }
  }
}

// ---------------- reductions (blockDim == 256) ----------------
__device__ __forceinline__ float block_sum256(float v, float* red) {
  const int t = threadIdx.x;
  red[t] = v; __syncthreads();
  #pragma unroll
  for (int off = 128; off > 0; off >>= 1) {
    if (t < off) red[t] += red[t + off];
    __syncthreads();
  }
  float r = red[0]; __syncthreads();
  return r;
}

__device__ __forceinline__ float block_max256(float v, float* red) {
  const int t = threadIdx.x;
  red[t] = v; __syncthreads();
  #pragma unroll
  for (int off = 128; off > 0; off >>= 1) {
    if (t < off) red[t] = fmaxf(red[t], red[t + off]);
    __syncthreads();
  }
  float r = red[0]; __syncthreads();
  return r;
}

// ---------------- concat + LayerNorm over KV=960 ----------------
__global__ __launch_bounds__(256)
void concat_ln_kernel(const float* __restrict__ e1, const float* __restrict__ e2,
                      const float* __restrict__ e3, const float* __restrict__ e4,
                      const float* __restrict__ g,  const float* __restrict__ b,
                      float* __restrict__ y) {
  __shared__ float red[256];
  const long row = blockIdx.x;   // b*N + n
  auto fetch = [&](int c) -> float {
    if (c < 64)  return e1[row * 64  + c];
    if (c < 192) return e2[row * 128 + (c - 64)];
    if (c < 448) return e3[row * 256 + (c - 192)];
    return e4[row * 512 + (c - 448)];
  };
  float s = 0.0f, s2 = 0.0f;
  for (int c = threadIdx.x; c < kKV; c += 256) {
    const float v = fetch(c); s += v; s2 += v * v;
  }
  s  = block_sum256(s,  red);
  s2 = block_sum256(s2, red);
  const float mean = s / (float)kKV;
  const float rinv = rsqrtf(s2 / (float)kKV - mean * mean + 1e-6f);
  for (int c = threadIdx.x; c < kKV; c += 256) {
    const float v = fetch(c);
    y[row * kKV + c] = (v - mean) * rinv * g[c] + b[c];
  }
}

// ---------------- generic LayerNorm over last dim C ----------------
__global__ __launch_bounds__(256)
void ln_kernel(const float* __restrict__ x, const float* __restrict__ g,
               const float* __restrict__ b, float* __restrict__ y, int C) {
  __shared__ float red[256];
  const long row = blockIdx.x;
  const float* xr = x + row * C;
  float* yr = y + row * C;
  float s = 0.0f, s2 = 0.0f;
  for (int c = threadIdx.x; c < C; c += 256) {
    const float v = xr[c]; s += v; s2 += v * v;
  }
  s  = block_sum256(s,  red);
  s2 = block_sum256(s2, red);
  const float mean = s / (float)C;
  const float rinv = rsqrtf(s2 / (float)C - mean * mean + 1e-6f);
  for (int c = threadIdx.x; c < C; c += 256)
    yr[c] = (xr[c] - mean) * rinv * g[c] + b[c];
}

// ---------------- InstanceNorm stats per (b,h): over C*KV ----------------
__global__ __launch_bounds__(256)
void inorm_stats_kernel(const float* __restrict__ s, float* __restrict__ stats,
                        long total) {
  __shared__ float red[256];
  const long base = (long)blockIdx.x * total;
  float a = 0.0f, a2 = 0.0f;
  for (long i = threadIdx.x; i < total; i += 256) {
    const float v = s[base + i]; a += v; a2 += v * v;
  }
  a  = block_sum256(a,  red);
  a2 = block_sum256(a2, red);
  if (threadIdx.x == 0) {
    const float m = a / (float)total;
    stats[blockIdx.x * 2 + 0] = m;
    stats[blockIdx.x * 2 + 1] = rsqrtf(a2 / (float)total - m * m + 1e-5f);
  }
}

// ---------------- instance-norm + softmax over KV (in place) ----------------
__global__ __launch_bounds__(256)
void inorm_softmax_kernel(float* __restrict__ s, const float* __restrict__ stats,
                          int C) {
  __shared__ float red[256];
  const long row = blockIdx.x;           // (b*H + h)*C + d
  const int  bh  = (int)(row / C);
  const float m    = stats[bh * 2 + 0];
  const float rinv = stats[bh * 2 + 1];
  float* sr = s + row * kKV;
  float mx = -3.4e38f;
  for (int j = threadIdx.x; j < kKV; j += 256)
    mx = fmaxf(mx, (sr[j] - m) * rinv);
  mx = block_max256(mx, red);
  float se = 0.0f;
  for (int j = threadIdx.x; j < kKV; j += 256)
    se += expf((sr[j] - m) * rinv - mx);
  se = block_sum256(se, red);
  const float inv = 1.0f / se;
  for (int j = threadIdx.x; j < kKV; j += 256)
    sr[j] = expf((sr[j] - m) * rinv - mx) * inv;
}

// ---------------- host orchestration ----------------
static void launch_gemm(hipStream_t s, const GemmP& p, int nz) {
  dim3 grid((unsigned)((p.N + BN - 1) / BN),
            (unsigned)((p.M + BM - 1) / BM),
            (unsigned)nz);
  gemm_bf16_wmma<<<grid, dim3(128), 0, s>>>(p);
}

extern "C" void kernel_launch(void* const* d_in, const int* in_sizes, int n_in,
                              void* d_out, int out_size, void* d_ws, size_t ws_size,
                              hipStream_t stream) {
  (void)in_sizes; (void)n_in; (void)out_size; (void)ws_size;

  // ---- input pointers (setup_inputs dict order) ----
  const float *emb[4], *ang[4], *anb[4], *Wq[4], *Wo[4], *fng[4], *fnb[4];
  const float *w1[4], *b1[4], *w2[4], *b2[4];
  for (int i = 0; i < 4; ++i) {
    const int base = i * 11;
    emb[i] = (const float*)d_in[base + 0];
    ang[i] = (const float*)d_in[base + 1];
    anb[i] = (const float*)d_in[base + 2];
    Wq[i]  = (const float*)d_in[base + 3];
    Wo[i]  = (const float*)d_in[base + 4];
    fng[i] = (const float*)d_in[base + 5];
    fnb[i] = (const float*)d_in[base + 6];
    w1[i]  = (const float*)d_in[base + 7];
    b1[i]  = (const float*)d_in[base + 8];
    w2[i]  = (const float*)d_in[base + 9];
    b2[i]  = (const float*)d_in[base + 10];
  }
  const float* anAg = (const float*)d_in[44];
  const float* anAb = (const float*)d_in[45];
  const float* Wk   = (const float*)d_in[46];
  const float* Wv   = (const float*)d_in[47];

  // ---- workspace layout (fp32) ----
  float* ws = (float*)d_ws;
  size_t o = 0;
  float* ea  = ws + o; o += (size_t)kB * kN * kKV;            // [B,N,KV]
  float* Kt  = ws + o; o += (size_t)kB * kH * kKV * kNp;      // [B,H,KV,Np] (transposed, token-padded)
  float* Vb  = ws + o; o += (size_t)kB * kH * kN * kKV;       // [B,H,N,KV]
  float* Qt  = ws + o; o += (size_t)kB * kH * 512 * kNp;      // [B,H,Cmax,Np] (transposed)
  float* cx  = ws + o; o += (size_t)kB * kN * 512;            // [B,N,Cmax]
  float* sc  = ws + o; o += (size_t)kB * kH * 512 * kKV;      // [B,H,Cmax,KV]
  float* st  = ws + o; o += 64;                               // stats [B*H*2]
  float* ctx = ws + o; o += (size_t)kB * kN * 512;            // [B,N,Cmax]
  float* xb  = ws + o; o += (size_t)kB * kN * 512;            // residual x
  float* hb  = ws + o; o += (size_t)kB * kN * 512;            // LN(x)
  float* ff  = ws + o; o += (size_t)kB * kN * 2048;           // FFN hidden

  const float scale = 1.0f / sqrtf((float)kKV);

  // 1) ea = LayerNorm(concat(embs))
  concat_ln_kernel<<<dim3(kB * kN), dim3(256), 0, stream>>>(
      emb[0], emb[1], emb[2], emb[3], anAg, anAb, ea);

  // 2) Kt[b,h,j,tok] = (ea[b] @ Wk[h]^T)^T ; V[b,h,tok,j] normal.  z = b*H + h
  {
    GemmP p{};                              // K projection, transposed store
    p.A = ea; p.Bm = Wk; p.bias = nullptr; p.C = Kt;
    p.M = kN; p.N = kKV; p.K = kKV; p.Mzero = kNp;
    p.a_rs = kKV; p.b_rs = kKV;
    p.c_rs = 1; p.c_cs = kNp;               // Kt[n, m]
    p.a_sb = (long)kN * kKV;       p.a_sh = 0;
    p.b_sb = 0;                    p.b_sh = (long)kKV * kKV;
    p.c_sb = (long)kH * kKV * kNp; p.c_sh = (long)kKV * kNp;
    p.H2 = kH; p.alpha = 1.0f; p.flags = 0;
    launch_gemm(stream, p, kB * kH);

    GemmP q = p;                            // V projection, normal store
    q.Bm = Wv; q.C = Vb; q.Mzero = kN;
    q.c_rs = kKV; q.c_cs = 1;
    q.c_sb = (long)kH * kN * kKV;  q.c_sh = (long)kN * kKV;
    launch_gemm(stream, q, kB * kH);
  }

  // 3) per-branch pipeline
  size_t out_off = 0;
  for (int i = 0; i < 4; ++i) {
    const int C  = kCs[i];
    const int C4 = 4 * C;

    // cx = LN(emb_i)
    ln_kernel<<<dim3(kB * kN), dim3(256), 0, stream>>>(emb[i], ang[i], anb[i], cx, C);

    // Qt[b,h,d,tok] = (cx[b] @ Wq[h]^T)^T   (token-padded with zeros)
    {
      GemmP p{};
      p.A = cx; p.Bm = Wq[i]; p.bias = nullptr; p.C = Qt;
      p.M = kN; p.N = C; p.K = C; p.Mzero = kNp;
      p.a_rs = C; p.b_rs = C;
      p.c_rs = 1; p.c_cs = kNp;             // Qt[n, m]
      p.a_sb = (long)kN * C;       p.a_sh = 0;
      p.b_sb = 0;                  p.b_sh = (long)C * C;
      p.c_sb = (long)kH * C * kNp; p.c_sh = (long)C * kNp;
      p.H2 = kH; p.alpha = 1.0f; p.flags = 0;
      launch_gemm(stream, p, kB * kH);
    }

    // scores[b,h] = scale * Qt[b,h] @ Kt[b,h]^T  (contiguous reduce over padded tokens)
    {
      GemmP p{};
      p.A = Qt; p.Bm = Kt; p.bias = nullptr; p.C = sc;
      p.M = C; p.N = kKV; p.K = kNp; p.Mzero = C;
      p.a_rs = kNp; p.b_rs = kNp;
      p.c_rs = kKV; p.c_cs = 1;
      p.a_sb = (long)kH * C * kNp;   p.a_sh = (long)C * kNp;
      p.b_sb = (long)kH * kKV * kNp; p.b_sh = (long)kKV * kNp;
      p.c_sb = (long)kH * C * kKV;   p.c_sh = (long)C * kKV;
      p.H2 = kH; p.alpha = scale; p.flags = 0;
      launch_gemm(stream, p, kB * kH);
    }

    // InstanceNorm stats per (b,h) over C*KV, then softmax over KV in place
    inorm_stats_kernel<<<dim3(kB * kH), dim3(256), 0, stream>>>(sc, st, (long)C * kKV);
    inorm_softmax_kernel<<<dim3(kB * kH * C), dim3(256), 0, stream>>>(sc, st, C);

    // ctx[b] = (1/H) * sum_h V[b,h] @ probs[b,h]^T     (accumulate over heads)
    for (int h = 0; h < kH; ++h) {
      GemmP p{};
      p.A = Vb + (long)h * kN * kKV;
      p.Bm = sc + (long)h * C * kKV;
      p.bias = nullptr; p.C = ctx;
      p.M = kN; p.N = C; p.K = kKV; p.Mzero = kN;
      p.a_rs = kKV; p.b_rs = kKV;
      p.c_rs = C; p.c_cs = 1;
      p.a_sb = (long)kH * kN * kKV; p.a_sh = 0;
      p.b_sb = (long)kH * C * kKV;  p.b_sh = 0;
      p.c_sb = (long)kN * C;        p.c_sh = 0;
      p.H2 = 1; p.alpha = 1.0f / (float)kH;
      p.flags = (h == 0) ? 0 : 1;   // first head stores, rest accumulate
      launch_gemm(stream, p, kB);
    }

    // x = emb_i + ctx @ Wo^T
    hipMemcpyAsync(xb, emb[i], (size_t)kB * kN * C * sizeof(float),
                   hipMemcpyDeviceToDevice, stream);
    {
      GemmP p{};
      p.A = ctx; p.Bm = Wo[i]; p.bias = nullptr; p.C = xb;
      p.M = kB * kN; p.N = C; p.K = C; p.Mzero = kB * kN;
      p.a_rs = C; p.b_rs = C; p.c_rs = C; p.c_cs = 1;
      p.a_sb = p.a_sh = p.b_sb = p.b_sh = p.c_sb = p.c_sh = 0;
      p.H2 = 1; p.alpha = 1.0f; p.flags = 1;  // accumulate onto residual
      launch_gemm(stream, p, 1);
    }

    // h = LN(x)
    ln_kernel<<<dim3(kB * kN), dim3(256), 0, stream>>>(xb, fng[i], fnb[i], hb, C);

    // ff = gelu(h @ w1^T + b1)
    {
      GemmP p{};
      p.A = hb; p.Bm = w1[i]; p.bias = b1[i]; p.C = ff;
      p.M = kB * kN; p.N = C4; p.K = C; p.Mzero = kB * kN;
      p.a_rs = C; p.b_rs = C; p.c_rs = C4; p.c_cs = 1;
      p.a_sb = p.a_sh = p.b_sb = p.b_sh = p.c_sb = p.c_sh = 0;
      p.H2 = 1; p.alpha = 1.0f; p.flags = 2 | 4;  // bias + gelu
      launch_gemm(stream, p, 1);
    }

    // out_i = x + (ff @ w2^T + b2)
    float* outp = (float*)d_out + out_off;
    hipMemcpyAsync(outp, xb, (size_t)kB * kN * C * sizeof(float),
                   hipMemcpyDeviceToDevice, stream);
    {
      GemmP p{};
      p.A = ff; p.Bm = w2[i]; p.bias = b2[i]; p.C = outp;
      p.M = kB * kN; p.N = C; p.K = C4; p.Mzero = kB * kN;
      p.a_rs = C4; p.b_rs = C4; p.c_rs = C; p.c_cs = 1;
      p.a_sb = p.a_sh = p.b_sb = p.b_sh = p.c_sb = p.c_sh = 0;
      p.H2 = 1; p.alpha = 1.0f; p.flags = 1 | 2;  // bias + accumulate residual
      launch_gemm(stream, p, 1);
    }

    out_off += (size_t)kB * kN * C;
  }
}
